// MultiAttentionLayer_53781580480695
// MI455X (gfx1250) — matrix operand order repaired and checked by reading, not compiled, not run on hardware
//
#include <hip/hip_runtime.h>
#include <hip/hip_bf16.h>

#define B_  4
#define F_  2048
#define T_  2048
#define H_  16
#define D_  64
#define HID_ 1024
#define ND_ 1024   // H*D

typedef __attribute__((ext_vector_type(16))) __bf16          v16bf;
typedef __attribute__((ext_vector_type(8)))  __bf16          bf16x8;
typedef __attribute__((ext_vector_type(8)))  float           v8f;
typedef __attribute__((ext_vector_type(8)))  float           f32x8;
typedef __attribute__((ext_vector_type(4)))  float           f32x4;
typedef __attribute__((ext_vector_type(8)))  unsigned short  u16x8;
typedef __attribute__((ext_vector_type(16))) unsigned short  u16x16;

// ---- hardware f32 -> bf16 converts ----
static __device__ __forceinline__ unsigned short f2bf1(float f) {
    return __builtin_bit_cast(unsigned short, (__bf16)f);
}
static __device__ __forceinline__ u16x8 cvt8(f32x4 a, f32x4 b) {
    f32x8 f = __builtin_shufflevector(a, b, 0,1,2,3,4,5,6,7);
    bf16x8 h = __builtin_convertvector(f, bf16x8);
    return __builtin_bit_cast(u16x8, h);
}
static __device__ __forceinline__ v16bf mkfrag(u16x8 lo, u16x8 hi) {
    u16x16 c = __builtin_shufflevector(lo, hi, 0,1,2,3,4,5,6,7,8,9,10,11,12,13,14,15);
    return __builtin_bit_cast(v16bf, c);
}
static __device__ __forceinline__ v8f wmma_bf16(v16bf a, v16bf b, v8f c) {
    return __builtin_amdgcn_wmma_f32_16x16x32_bf16(false, a, false, b, (short)0, c, false, false);
}

// ---- CDNA5 async global->LDS copies (ASYNCcnt-tracked) ----
#define ASYNC_B128(ldsoff, gptr)                                               \
    asm volatile("global_load_async_to_lds_b128 %0, %1, off"                   \
                 :: "v"(ldsoff), "v"(gptr) : "memory")
#define ASYNC_B32(ldsoff, gptr)                                                \
    asm volatile("global_load_async_to_lds_b32 %0, %1, off"                    \
                 :: "v"(ldsoff), "v"(gptr) : "memory")
#define WAIT_ASYNC() asm volatile("s_wait_asynccnt 0x0" ::: "memory")

static __device__ __forceinline__ unsigned lds_off(const void* p) {
    return (unsigned)(uintptr_t)p;   // flat->LDS: hw truncates to low 32 bits
}

// ---------------------------------------------------------------------------
// Kernel 1: QKV projection GEMM.  C[M,1024] = X[M,1024] * W[1024,1024] + bias
// blockIdx.z: 0 -> Q, 1 -> K, 2 -> V (transposed output)
// WG 256 thr = 8 waves; tile 128(M) x 64(N); wave 32x32 (2x2 WMMA); K step 32.
// Tiles staged fp32 via async-LDS (W transposed during the async scatter).
// ---------------------------------------------------------------------------
__global__ __launch_bounds__(256)
void qkv_proj_kernel(const float* __restrict__ from_t, const float* __restrict__ to_t,
                     const float* __restrict__ Wq, const float* __restrict__ bq,
                     const float* __restrict__ Wk, const float* __restrict__ bk,
                     const float* __restrict__ Wv, const float* __restrict__ bv,
                     unsigned short* __restrict__ Qb, unsigned short* __restrict__ Kb,
                     unsigned short* __restrict__ Vtb)
{
    __shared__ float Asf[128 * 36];   // A tile fp32 [m][k], stride 36
    __shared__ float Bsf[64 * 36];    // W tile fp32 TRANSPOSED [n][k], stride 36

    const float* X;  const float* W;  const float* bias;
    const int gz = blockIdx.z;
    if (gz == 0)      { X = from_t; W = Wq; bias = bq; }
    else if (gz == 1) { X = to_t;   W = Wk; bias = bk; }
    else              { X = to_t;   W = Wv; bias = bv; }

    const int tid  = threadIdx.x;
    const int lane = tid & 31;
    const int wave = tid >> 5;
    const int lr   = lane & 15;
    const int hi   = lane >> 4;

    const int m0 = blockIdx.y * 128;
    const int n0 = blockIdx.x * 64;
    const int wm = (wave & 3) * 32;
    const int wn = (wave >> 2) * 32;

    v8f acc[2][2];
    acc[0][0] = (v8f){}; acc[0][1] = (v8f){}; acc[1][0] = (v8f){}; acc[1][1] = (v8f){};

    for (int kt = 0; kt < HID_; kt += 32) {
        // ---- async stage A tile: 128x32 fp32, b128 chunks ----
#pragma unroll
        for (int i = 0; i < 4; i++) {
            int idx = tid + 256 * i;
            int r = idx >> 3, c = idx & 7;
            const float* g = X + (size_t)(m0 + r) * HID_ + kt + c * 4;
            ASYNC_B128(lds_off(&Asf[r * 36 + c * 4]), g);
        }
        // ---- async stage W tile TRANSPOSED: per-lane b32 scatter ----
#pragma unroll
        for (int i = 0; i < 8; i++) {
            int idx = tid + 256 * i;
            int k = idx >> 6, n = idx & 63;
            const float* g = W + (size_t)(kt + k) * ND_ + n0 + n;
            ASYNC_B32(lds_off(&Bsf[n * 36 + k]), g);
        }
        WAIT_ASYNC();
        __syncthreads();

        // ---- fragments (fp32 LDS -> hw cvt -> bf16 frags) + 4 WMMAs ----
        v16bf af[2], bf[2];
#pragma unroll
        for (int mi = 0; mi < 2; mi++) {
            const float* ap = &Asf[(wm + mi * 16 + lr) * 36];
            u16x8 lo = cvt8(*(const f32x4*)(ap + 8 * hi),
                            *(const f32x4*)(ap + 8 * hi + 4));
            u16x8 hi8 = cvt8(*(const f32x4*)(ap + 16 + 8 * hi),
                             *(const f32x4*)(ap + 16 + 8 * hi + 4));
            af[mi] = mkfrag(lo, hi8);
        }
#pragma unroll
        for (int ni = 0; ni < 2; ni++) {
            const float* bp = &Bsf[(wn + ni * 16 + lr) * 36 + 8 * hi];
            u16x8 lo = cvt8(*(const f32x4*)(bp),      *(const f32x4*)(bp + 4));
            u16x8 hi8 = cvt8(*(const f32x4*)(bp + 16), *(const f32x4*)(bp + 20));
            bf[ni] = mkfrag(lo, hi8);
        }
#pragma unroll
        for (int mi = 0; mi < 2; mi++)
#pragma unroll
            for (int ni = 0; ni < 2; ni++)
                acc[mi][ni] = wmma_bf16(af[mi], bf[ni], acc[mi][ni]);
        __syncthreads();
    }

    // ---- epilogue: bias add + bf16 store to head-major layouts ----
#pragma unroll
    for (int mi = 0; mi < 2; mi++) {
#pragma unroll
        for (int ni = 0; ni < 2; ni++) {
            int ng = n0 + wn + ni * 16 + lr;
            float bval = bias[ng];
            int hh = ng >> 6, dd = ng & 63;
            int mg0 = m0 + wm + mi * 16 + 8 * hi;
            if (gz == 2) {
                int bb = mg0 >> 11, tt = mg0 & 2047;
                u16x8 pk;
#pragma unroll
                for (int j = 0; j < 8; j++) pk[j] = f2bf1(acc[mi][ni][j] + bval);
                *(u16x8*)(Vtb + ((size_t)(bb * H_ + hh) * D_ + dd) * T_ + tt) = pk;
            } else {
                unsigned short* o = (gz == 0) ? Qb : Kb;
#pragma unroll
                for (int j = 0; j < 8; j++) {
                    int mg = mg0 + j;
                    int bb = mg >> 11, ff = mg & 2047;
                    o[((size_t)(bb * H_ + hh) * F_ + ff) * D_ + dd]
                        = f2bf1(acc[mi][ni][j] + bval);
                }
            }
        }
    }
}

// ---------------------------------------------------------------------------
// Kernel 2: flash attention. WG = (b,h, 128 F rows), 8 waves x 16 query rows.
// T streamed in tiles of 64; K/V tiles staged into LDS via async copies and
// shared by all 8 waves.
// ---------------------------------------------------------------------------
__global__ __launch_bounds__(256)
void attn_kernel(const unsigned short* __restrict__ Qb,
                 const unsigned short* __restrict__ Kb,
                 const unsigned short* __restrict__ Vtb,
                 const int* __restrict__ mask,
                 float* __restrict__ out)
{
    __shared__ unsigned short Ks[64 * 72];        // K tile  [t][d], stride 72
    __shared__ unsigned short Vs[64 * 72];        // Vt tile [d][t], stride 72
    __shared__ unsigned short pbuf[8][16 * 72];   // per-wave P transpose staging

    const int bh = blockIdx.y;
    const int b  = bh >> 4;
    const int h  = bh & 15;
    const int tid  = threadIdx.x;
    const int lane = tid & 31;
    const int wave = tid >> 5;
    const int lr = lane & 15;
    const int hi = lane >> 4;
    const int f0 = blockIdx.x * 128 + wave * 16;

    // Resident Q fragments (K-dim 0..31, 32..63)
    v16bf qf[2];
    {
        const unsigned short* qp = Qb + ((size_t)bh * F_ + f0 + lr) * D_;
#pragma unroll
        for (int kk = 0; kk < 2; kk++)
            qf[kk] = mkfrag(*(const u16x8*)(qp + kk * 32 + 16 * hi),
                            *(const u16x8*)(qp + kk * 32 + 16 * hi + 8));
    }

    float mrow[8], lrow[8];
    v8f ctxa[4];
#pragma unroll
    for (int j = 0; j < 8; j++) { mrow[j] = -3.0e38f; lrow[j] = 0.0f; }
#pragma unroll
    for (int d2 = 0; d2 < 4; d2++) ctxa[d2] = (v8f){};

    for (int t0 = 0; t0 < T_; t0 += 64) {
        // ---- async stage K tile (64x64) and Vt tile (64x64) into LDS ----
#pragma unroll
        for (int i = 0; i < 2; i++) {
            int idx = tid + 256 * i;
            int r = idx >> 3, c = idx & 7;
            const unsigned short* gk = Kb + ((size_t)bh * T_ + t0 + r) * D_ + c * 8;
            ASYNC_B128(lds_off(&Ks[r * 72 + c * 8]), gk);
            const unsigned short* gv = Vtb + ((size_t)bh * D_ + r) * T_ + t0 + c * 8;
            ASYNC_B128(lds_off(&Vs[r * 72 + c * 8]), gv);
        }
        WAIT_ASYNC();
        __syncthreads();

        // ---- S = Q * K^T : 16x64 tile = four 16x16 WMMA columns ----
        v8f s[4];
#pragma unroll
        for (int u = 0; u < 4; u++) {
            const unsigned short* kp = &Ks[(u * 16 + lr) * 72];
            v16bf kf0 = mkfrag(*(const u16x8*)(kp + 16 * hi),
                               *(const u16x8*)(kp + 16 * hi + 8));
            v16bf kf1 = mkfrag(*(const u16x8*)(kp + 32 + 16 * hi),
                               *(const u16x8*)(kp + 32 + 16 * hi + 8));
            v8f a = (v8f){};
            a = wmma_bf16(qf[0], kf0, a);
            a = wmma_bf16(qf[1], kf1, a);
            s[u] = a;
        }

        // ---- scale + additive mask ----
#pragma unroll
        for (int u = 0; u < 4; u++) {
            const int* mp = mask + ((size_t)b * F_ + f0 + 8 * hi) * (size_t)T_ + t0 + u * 16 + lr;
#pragma unroll
            for (int j = 0; j < 8; j++) {
                float mv = (float)mp[(size_t)j * T_];
                s[u][j] = s[u][j] * 0.125f + (1.0f - mv) * (-10000.0f);
            }
        }

        // ---- online softmax (rows live across 16-lane halves) ----
#pragma unroll
        for (int j = 0; j < 8; j++) {
            float vmax = fmaxf(fmaxf(s[0][j], s[1][j]), fmaxf(s[2][j], s[3][j]));
            vmax = fmaxf(vmax, __shfl_xor(vmax, 1));
            vmax = fmaxf(vmax, __shfl_xor(vmax, 2));
            vmax = fmaxf(vmax, __shfl_xor(vmax, 4));
            vmax = fmaxf(vmax, __shfl_xor(vmax, 8));
            float mnew = fmaxf(mrow[j], vmax);
            float sc = exp2f((mrow[j] - mnew) * 1.44269504f);
            float p0 = exp2f((s[0][j] - mnew) * 1.44269504f);
            float p1 = exp2f((s[1][j] - mnew) * 1.44269504f);
            float p2 = exp2f((s[2][j] - mnew) * 1.44269504f);
            float p3 = exp2f((s[3][j] - mnew) * 1.44269504f);
            float ps = (p0 + p1) + (p2 + p3);
            ps += __shfl_xor(ps, 1);
            ps += __shfl_xor(ps, 2);
            ps += __shfl_xor(ps, 4);
            ps += __shfl_xor(ps, 8);
            lrow[j] = lrow[j] * sc + ps;
            mrow[j] = mnew;
            ctxa[0][j] *= sc; ctxa[1][j] *= sc; ctxa[2][j] *= sc; ctxa[3][j] *= sc;
            unsigned short* pw = &pbuf[wave][(j + 8 * hi) * 72 + lr];
            pw[0]  = f2bf1(p0);
            pw[16] = f2bf1(p1);
            pw[32] = f2bf1(p2);
            pw[48] = f2bf1(p3);
        }

        // ---- reload P as two A-fragments (same-wave LDS ops are in-order) ----
        const unsigned short* pp = &pbuf[wave][lr * 72];
        v16bf pf[2];
#pragma unroll
        for (int kk = 0; kk < 2; kk++)
            pf[kk] = mkfrag(*(const u16x8*)(pp + kk * 32 + 16 * hi),
                            *(const u16x8*)(pp + kk * 32 + 16 * hi + 8));

        // ---- ctx += P * V (K-dim 64 = two 32 chunks) ----
#pragma unroll
        for (int d2 = 0; d2 < 4; d2++) {
            const unsigned short* vp = &Vs[(d2 * 16 + lr) * 72];
            v16bf vf0 = mkfrag(*(const u16x8*)(vp + 16 * hi),
                               *(const u16x8*)(vp + 16 * hi + 8));
            v16bf vf1 = mkfrag(*(const u16x8*)(vp + 32 + 16 * hi),
                               *(const u16x8*)(vp + 32 + 16 * hi + 8));
            ctxa[d2] = wmma_bf16(pf[0], vf0, ctxa[d2]);
            ctxa[d2] = wmma_bf16(pf[1], vf1, ctxa[d2]);
        }
        __syncthreads();   // before next tile overwrites Ks/Vs
    }

    // ---- finalize: divide by row sum, write [B,F,H*D] fp32 ----
#pragma unroll
    for (int d2 = 0; d2 < 4; d2++) {
#pragma unroll
        for (int j = 0; j < 8; j++) {
            int f = f0 + j + 8 * hi;
            out[((size_t)b * F_ + f) * ND_ + h * D_ + d2 * 16 + lr] = ctxa[d2][j] / lrow[j];
        }
    }
}

// ---------------------------------------------------------------------------
extern "C" void kernel_launch(void* const* d_in, const int* in_sizes, int n_in,
                              void* d_out, int out_size, void* d_ws, size_t ws_size,
                              hipStream_t stream) {
    (void)in_sizes; (void)n_in; (void)out_size; (void)ws_size;
    const float* from_t = (const float*)d_in[0];
    const float* to_t   = (const float*)d_in[1];
    const int*   mask   = (const int*)d_in[2];
    const float* Wq = (const float*)d_in[3];
    const float* bq = (const float*)d_in[4];
    const float* Wk = (const float*)d_in[5];
    const float* bk = (const float*)d_in[6];
    const float* Wv = (const float*)d_in[7];
    const float* bv = (const float*)d_in[8];
    float* out = (float*)d_out;

    const size_t elems = (size_t)B_ * H_ * F_ * D_;
    unsigned short* Qb  = (unsigned short*)d_ws;
    unsigned short* Kb  = Qb + elems;
    unsigned short* Vtb = Kb + elems;

    dim3 g1(ND_ / 64, (B_ * F_) / 128, 3);
    qkv_proj_kernel<<<g1, 256, 0, stream>>>(from_t, to_t, Wq, bq, Wk, bk, Wv, bv,
                                            Qb, Kb, Vtb);

    dim3 g2(F_ / 128, B_ * H_);
    attn_kernel<<<g2, 256, 0, stream>>>(Qb, Kb, Vtb, mask, out);
}
